// HSMSSD_747324310039
// MI455X (gfx1250) — matrix operand order repaired and checked by reading, not compiled, not run on hardware
//
#include <hip/hip_runtime.h>
#include <stdint.h>

// ---------------------------------------------------------------------------
// HSM-SSD block for MI455X (gfx1250, wave32, WMMA, async global->LDS).
// Pipeline:
//   1) k_ln        : per-pixel channel mean/rstd + x -> bf16 copy
//   2) k_bcdt_gemm : bcdt = W_bcdt @ LN(x)   (bf16 WMMA, LN fused in B-tile)
//   3) k_gconv     : 3x3 grouped conv (groups of 4), bf16 in/out
//   4) k_softmax_ab: AB = softmax_N(dt + A) * BC   (bf16 out)
//   5) k_h_gemm    : h = Xbf @ ABbf^T  (K=4096, bf16 WMMA,
//                    double-buffered global_load_async_to_lds_b64 pipeline)
//   6) k_h2        : h2 = W_out @ h    (tiny GEMM, fp32 VALU, bf16 out)
//   7) k_y_out     : out = h2 @ BC + x (bf16 WMMA, fused residual)
// Algebraic trick: W_out @ (h @ BC) == (W_out @ h) @ BC  -> kills the
// 17 GFLOP out-proj GEMM and 128MB of intermediate traffic.
// ---------------------------------------------------------------------------

#define DEV __device__ __forceinline__

constexpr int B_  = 32;
constexpr int C_  = 256;
constexpr int S_  = 64;
constexpr int CD_ = 128;
constexpr int N_  = 4096;   // 64*64
constexpr float EPSV = 1e-5f;

typedef unsigned short u16;
typedef unsigned int   u32;
typedef __bf16 bf16_t;
typedef bf16_t v16bf __attribute__((ext_vector_type(16)));
typedef float  v8f   __attribute__((ext_vector_type(8)));
typedef u32    v8u   __attribute__((ext_vector_type(8)));

union FragU { v8u u; v16bf b; };

// ---- bf16 helpers (manual, RNE) -------------------------------------------
DEV u16 f2bf(float f) {
  u32 u = __float_as_uint(f);
  u32 r = (u + 0x7FFFu + ((u >> 16) & 1u)) >> 16;
  return (u16)r;
}
DEV float bf2f(u16 h) { return __uint_as_float(((u32)h) << 16); }

// ---- WMMA fragment builders (contiguous k order, same in A and B, wave32) ------------------
// A (16x32 bf16): lane L<16 -> row L, K 0..15; L>=16 -> row L-16,
// K 16..31. LDS tile is row-major [m][k], ld in u16 units (even).
DEV v16bf frag_a(const u16* lds, int ld, int m_off, int lane) {
  const u16* p = lds + (size_t)(m_off + (lane & 15)) * ld;
  const int kb = (lane & 16) ? 16 : 0;
  FragU f;
#pragma unroll
  for (int i = 0; i < 8; ++i) f.u[i] = *(const u32*)(p + kb + 2 * i);
  return f.b;
}
// B (32x16 bf16): lane L<16 -> col L, K 0..15; L>=16 -> col L-16, K 16..31.
// LDS tile stored transposed [n][k] so each lane reads 8 contiguous dwords.
DEV v16bf frag_b(const u16* lds, int ld, int n_off, int lane) {
  const u16* p = lds + (size_t)(n_off + (lane & 15)) * ld + ((lane & 16) ? 16 : 0);
  FragU f;
#pragma unroll
  for (int i = 0; i < 8; ++i) f.u[i] = *(const u32*)(p + 2 * i);
  return f.b;
}

DEV v8f wmma_bf16(v16bf a, v16bf b, v8f c) {
  return __builtin_amdgcn_wmma_f32_16x16x32_bf16(false, a, false, b,
                                                 (short)0, c, false, false);
}

// D tile (16x16 f32): lane L -> col L&15, rows (L<16 ? i : 8+i) for vgpr i.
DEV void store_tile_f32(float* dst, int ldd, int lane, v8f a) {
  const int n = lane & 15, mb = (lane & 16) ? 8 : 0;
#pragma unroll
  for (int i = 0; i < 8; ++i) dst[(size_t)(mb + i) * ldd + n] = a[i];
}
DEV void store_tile_bf16(u16* dst, int ldd, int lane, v8f a) {
  const int n = lane & 15, mb = (lane & 16) ? 8 : 0;
#pragma unroll
  for (int i = 0; i < 8; ++i) dst[(size_t)(mb + i) * ldd + n] = f2bf(a[i]);
}
DEV void store_tile_res(float* dst, const float* res, int ldd, int lane, v8f a) {
  const int n = lane & 15, mb = (lane & 16) ? 8 : 0;
#pragma unroll
  for (int i = 0; i < 8; ++i)
    dst[(size_t)(mb + i) * ldd + n] = a[i] + res[(size_t)(mb + i) * ldd + n];
}

// ---- CDNA5 async global->LDS (tracked by ASYNCcnt) ------------------------
// GVS mode: mem = SADDR(64) + VADDR(32) ; VDST vgpr holds LDS byte address.
DEV void async_ld_b64(unsigned lds_byte_off, const void* base, unsigned gbyte_off) {
  asm volatile("global_load_async_to_lds_b64 %0, %1, %2"
               :: "v"(lds_byte_off), "v"(gbyte_off), "s"(base)
               : "memory");
}
DEV unsigned lds_off(const u16* p) {   // low 32 bits of LDS-aperture flat addr
  return (unsigned)(uintptr_t)p;
}

// ---------------------------------------------------------------------------
// 1) LayerNorm stats + bf16 copy of x.  grid: B*N/256 blocks x 256 thr.
__global__ void k_ln(const float* __restrict__ x, float* __restrict__ mean,
                     float* __restrict__ rstd, u16* __restrict__ xbf) {
  const int pix = blockIdx.x * blockDim.x + threadIdx.x;   // b*N + n
  const int b = pix >> 12;
  const int n = pix & (N_ - 1);
  const float* xp = x + (size_t)b * C_ * N_ + n;
  u16* xo = xbf + (size_t)b * C_ * N_ + n;
  float s = 0.f, ss = 0.f;
  for (int c = 0; c < C_; ++c) {
    float v = xp[(size_t)c * N_];
    s += v; ss += v * v;
    xo[(size_t)c * N_] = f2bf(v);
  }
  const float m = s * (1.f / (float)C_);
  const float var = (ss - (float)C_ * m * m) * (1.f / (float)(C_ - 1));
  mean[pix] = m;
  rstd[pix] = rsqrtf(var + EPSV);
}

// ---------------------------------------------------------------------------
// 2) bcdt = W_bcdt(128x256) @ xn(256x4096), LN fused into B-tile build.
//    grid: (N/64, CD/64, B), 256 thr = 8 waves, block tile 64x64.
__global__ void k_bcdt_gemm(const u16* __restrict__ xbf, const float* __restrict__ wb,
                            const float* __restrict__ mean, const float* __restrict__ rstd,
                            const float* __restrict__ lnw, const float* __restrict__ lnb,
                            u16* __restrict__ bcdt) {
  __shared__ u16 lA[64 * 36];
  __shared__ u16 lB[64 * 36];
  const int b = blockIdx.z, m0 = blockIdx.y * 64, n0 = blockIdx.x * 64;
  const int tid = threadIdx.x, lane = tid & 31, wave = tid >> 5;
  const int wm = (wave & 3) * 16, wn = (wave >> 2) * 32;
  const u16* xb = xbf + (size_t)b * C_ * N_;
  const float* mu = mean + (size_t)b * N_;
  const float* rs = rstd + (size_t)b * N_;
  v8f acc0 = {0, 0, 0, 0, 0, 0, 0, 0};
  v8f acc1 = acc0;
  for (int k0 = 0; k0 < C_; k0 += 32) {
#pragma unroll
    for (int j = 0; j < 8; ++j) {                      // A: 64(m) x 32(k)
      int idx = tid + j * 256;
      int r = idx >> 5, kk = idx & 31;
      lA[r * 36 + kk] = f2bf(wb[(size_t)(m0 + r) * C_ + k0 + kk]);
    }
#pragma unroll
    for (int j = 0; j < 8; ++j) {                      // B^T: [n][k], LN fused
      int idx = tid + j * 256;
      int kl = idx >> 6, nl = idx & 63;
      int c = k0 + kl, n = n0 + nl;
      float v = bf2f(xb[(size_t)c * N_ + n]);
      v = (v - mu[n]) * rs[n] * lnw[c] + lnb[c];
      lB[nl * 36 + kl] = f2bf(v);
    }
    __syncthreads();
    v16bf a  = frag_a(lA, 36, wm, lane);
    v16bf b0 = frag_b(lB, 36, wn, lane);
    v16bf b1 = frag_b(lB, 36, wn + 16, lane);
    acc0 = wmma_bf16(a, b0, acc0);
    acc1 = wmma_bf16(a, b1, acc1);
    __syncthreads();
  }
  u16* dp = bcdt + ((size_t)b * CD_ + m0) * N_ + n0;
  store_tile_bf16(dp + (size_t)wm * N_ + wn,      N_, lane, acc0);
  store_tile_bf16(dp + (size_t)wm * N_ + wn + 16, N_, lane, acc1);
}

// ---------------------------------------------------------------------------
// 3) 3x3 grouped conv, 4 ch/group, pad 1.  grid: (N/256, CD, B).
__global__ void k_gconv(const u16* __restrict__ in, const float* __restrict__ w,
                        u16* __restrict__ out) {
  const int n = blockIdx.x * blockDim.x + threadIdx.x;
  const int o = blockIdx.y, b = blockIdx.z;
  const int hy = n >> 6, wx = n & 63;
  const u16* ip = in + ((size_t)b * CD_ + (size_t)(o >> 2) * 4) * N_;
  const float* wp = w + (size_t)o * 36;   // (4 in-ch) x 3 x 3
  float acc = 0.f;
#pragma unroll
  for (int ci = 0; ci < 4; ++ci) {
    const u16* ic = ip + (size_t)ci * N_;
    const float* wc = wp + ci * 9;
#pragma unroll
    for (int dy = -1; dy <= 1; ++dy) {
      int yy = hy + dy;
      if ((unsigned)yy < 64u) {
#pragma unroll
        for (int dx = -1; dx <= 1; ++dx) {
          int xx = wx + dx;
          if ((unsigned)xx < 64u)
            acc += wc[(dy + 1) * 3 + (dx + 1)] * bf2f(ic[yy * 64 + xx]);
        }
      }
    }
  }
  out[((size_t)b * CD_ + o) * N_ + n] = f2bf(acc);
}

// ---------------------------------------------------------------------------
// 4) AB = softmax_N(dt + A[s]) * BC.  One block per (s,b) row of 4096.
__global__ void k_softmax_ab(const u16* __restrict__ bcdt2, const float* __restrict__ A,
                             u16* __restrict__ ab) {
  __shared__ float red[8];
  __shared__ float bcast;
  const int s = blockIdx.x, b = blockIdx.y;
  const int tid = threadIdx.x, lane = tid & 31, wave = tid >> 5;
  const u16* dtp = bcdt2 + ((size_t)b * CD_ + S_ + s) * N_;
  const u16* bcp = bcdt2 + ((size_t)b * CD_ + s) * N_;
  const float av = A[s];
  float v[16];
  float mx = -3.4e38f;
#pragma unroll
  for (int i = 0; i < 16; ++i) {
    v[i] = bf2f(dtp[tid + i * 256]) + av;
    mx = fmaxf(mx, v[i]);
  }
#pragma unroll
  for (int off = 16; off > 0; off >>= 1) mx = fmaxf(mx, __shfl_xor(mx, off, 32));
  if (lane == 0) red[wave] = mx;
  __syncthreads();
  if (tid == 0) {
    float m = red[0];
    for (int i = 1; i < 8; ++i) m = fmaxf(m, red[i]);
    bcast = m;
  }
  __syncthreads();
  mx = bcast;
  float sum = 0.f;
#pragma unroll
  for (int i = 0; i < 16; ++i) { v[i] = __expf(v[i] - mx); sum += v[i]; }
#pragma unroll
  for (int off = 16; off > 0; off >>= 1) sum += __shfl_xor(sum, off, 32);
  __syncthreads();
  if (lane == 0) red[wave] = sum;
  __syncthreads();
  if (tid == 0) {
    float t = 0.f;
    for (int i = 0; i < 8; ++i) t += red[i];
    bcast = 1.f / t;
  }
  __syncthreads();
  const float inv = bcast;
  u16* abp = ab + ((size_t)b * S_ + s) * N_;
#pragma unroll
  for (int i = 0; i < 16; ++i)
    abp[tid + i * 256] = f2bf(v[i] * inv * bf2f(bcp[tid + i * 256]));
}

// ---------------------------------------------------------------------------
// 5) h(256x64) = Xbf(256x4096) @ ABbf^T(4096x64), NT GEMM, K=4096.
//    grid: (C/64, B).  Double-buffered async global->LDS pipeline:
//    issue tile t+1 (4x async_b64/thread), s_wait_asynccnt 4 -> tile t ready.
__global__ void k_h_gemm(const u16* __restrict__ xbf, const u16* __restrict__ ab,
                         float* __restrict__ h) {
  __shared__ __align__(16) u16 lA[2][64 * 36];
  __shared__ __align__(16) u16 lB[2][64 * 36];
  const int b = blockIdx.y, m0 = blockIdx.x * 64;
  const int tid = threadIdx.x, lane = tid & 31, wave = tid >> 5;
  const int wm = (wave & 3) * 16, wn = (wave >> 2) * 32;
  const u16* Ag = xbf + ((size_t)b * C_ + m0) * N_;   // [c][n]
  const u16* Bg = ab + (size_t)b * S_ * N_;           // [s][n] == B^T storage

  // Per-thread async chunks: 64x32-half tile = 512 b64 chunks; 2 per thread.
  // chunk q: row r = q>>3, k-offset ko = (q&7)*4 halfs (8B aligned both sides).
  const int q0 = tid, q1 = tid + 256;
  const int r0 = q0 >> 3, ko0 = (q0 & 7) << 2;
  const int r1 = q1 >> 3, ko1 = (q1 & 7) << 2;
  const unsigned gA0 = (unsigned)((r0 * N_ + ko0) * 2);
  const unsigned gA1 = (unsigned)((r1 * N_ + ko1) * 2);
  unsigned la0[2], la1[2], lb0[2], lb1[2];
#pragma unroll
  for (int d = 0; d < 2; ++d) {
    la0[d] = lds_off(&lA[d][r0 * 36 + ko0]);
    la1[d] = lds_off(&lA[d][r1 * 36 + ko1]);
    lb0[d] = lds_off(&lB[d][r0 * 36 + ko0]);
    lb1[d] = lds_off(&lB[d][r1 * 36 + ko1]);
  }
  auto issue = [&](int d, int k0) {
    const unsigned kb = (unsigned)(k0 << 1);
    async_ld_b64(la0[d], Ag, gA0 + kb);
    async_ld_b64(la1[d], Ag, gA1 + kb);
    async_ld_b64(lb0[d], Bg, gA0 + kb);   // B tile has identical per-lane map
    async_ld_b64(lb1[d], Bg, gA1 + kb);
  };

  v8f acc0 = {0, 0, 0, 0, 0, 0, 0, 0};
  v8f acc1 = acc0;
  issue(0, 0);
  constexpr int NT = N_ / 32;
  for (int it = 0; it < NT; ++it) {
    const int cur = it & 1;
    if (it + 1 < NT) {
      issue(cur ^ 1, (it + 1) * 32);
      asm volatile("s_wait_asynccnt 0x4" ::: "memory");  // current 4 retired
    } else {
      asm volatile("s_wait_asynccnt 0x0" ::: "memory");
    }
    __syncthreads();                       // all waves' tile data visible
    v16bf a  = frag_a(lA[cur], 36, wm, lane);
    v16bf b0 = frag_b(lB[cur], 36, wn, lane);
    v16bf b1 = frag_b(lB[cur], 36, wn + 16, lane);
    acc0 = wmma_bf16(a, b0, acc0);
    acc1 = wmma_bf16(a, b1, acc1);
    __syncthreads();                       // done reading before next overwrite
  }
  float* hp = h + ((size_t)b * C_ + m0) * S_;
  store_tile_f32(hp + (size_t)wm * S_ + wn,      S_, lane, acc0);
  store_tile_f32(hp + (size_t)wm * S_ + wn + 16, S_, lane, acc1);
}

// ---------------------------------------------------------------------------
// 6) h2 = W_out(256x256) @ h(256x64), tiny -> fp32 VALU.  grid: (C/4, B).
__global__ void k_h2(const float* __restrict__ wout, const float* __restrict__ h,
                     u16* __restrict__ h2) {
  const int b = blockIdx.y;
  const int o = blockIdx.x * 4 + (threadIdx.x >> 6);
  const int s = threadIdx.x & 63;
  const float* wp = wout + (size_t)o * C_;
  const float* hp = h + (size_t)b * C_ * S_ + s;
  float acc = 0.f;
  for (int c = 0; c < C_; ++c) acc += wp[c] * hp[(size_t)c * S_];
  h2[((size_t)b * C_ + o) * S_ + s] = f2bf(acc);
}

// ---------------------------------------------------------------------------
// 7) out = h2(256x64) @ BC(64x4096) + x.  grid: (N/64, C/64, B), K=64.
__global__ void k_y_out(const u16* __restrict__ h2, const u16* __restrict__ bcdt2,
                        const float* __restrict__ x, float* __restrict__ out) {
  __shared__ u16 lA[64 * 36];
  __shared__ u16 lB[64 * 36];
  const int b = blockIdx.z, m0 = blockIdx.y * 64, n0 = blockIdx.x * 64;
  const int tid = threadIdx.x, lane = tid & 31, wave = tid >> 5;
  const int wm = (wave & 3) * 16, wn = (wave >> 2) * 32;
  const u16* Ag = h2 + ((size_t)b * C_ + m0) * S_;    // [o][s]
  const u16* Bg = bcdt2 + (size_t)b * CD_ * N_;       // channels 0..63 = BC
  v8f acc0 = {0, 0, 0, 0, 0, 0, 0, 0};
  v8f acc1 = acc0;
  for (int k0 = 0; k0 < S_; k0 += 32) {
#pragma unroll
    for (int j = 0; j < 4; ++j) {                     // A: contiguous dwords
      int q = tid + j * 256;
      int r = q >> 4, kk = (q & 15) << 1;
      *(u32*)&lA[r * 36 + kk] = *(const u32*)&Ag[(size_t)r * S_ + k0 + kk];
    }
#pragma unroll
    for (int j = 0; j < 8; ++j) {                     // B: transpose to [n][k]
      int idx = tid + j * 256;
      int kl = idx >> 6, nl = idx & 63;
      lB[nl * 36 + kl] = Bg[(size_t)(k0 + kl) * N_ + n0 + nl];
    }
    __syncthreads();
    v16bf a  = frag_a(lA, 36, wm, lane);
    v16bf b0 = frag_b(lB, 36, wn, lane);
    v16bf b1 = frag_b(lB, 36, wn + 16, lane);
    acc0 = wmma_bf16(a, b0, acc0);
    acc1 = wmma_bf16(a, b1, acc1);
    __syncthreads();
  }
  const float* xp = x + ((size_t)b * C_ + m0) * N_ + n0;
  float* op = out + ((size_t)b * C_ + m0) * N_ + n0;
  store_tile_res(op + (size_t)wm * N_ + wn, xp + (size_t)wm * N_ + wn, N_, lane, acc0);
  store_tile_res(op + (size_t)wm * N_ + wn + 16, xp + (size_t)wm * N_ + wn + 16, N_, lane, acc1);
}

// ---------------------------------------------------------------------------
// Workspace layout (bytes). Total ~148 MiB.
constexpr size_t OFF_XBF   = 0;                                      // 64 MiB
constexpr size_t OFF_BCDT  = OFF_XBF   + (size_t)B_ * C_ * N_ * 2;   // 32 MiB
constexpr size_t OFF_BCDT2 = OFF_BCDT  + (size_t)B_ * CD_ * N_ * 2;  // 32 MiB
constexpr size_t OFF_AB    = OFF_BCDT2 + (size_t)B_ * CD_ * N_ * 2;  // 16 MiB
constexpr size_t OFF_MEAN  = OFF_AB    + (size_t)B_ * S_ * N_ * 2;   // 512 KiB
constexpr size_t OFF_RSTD  = OFF_MEAN  + (size_t)B_ * N_ * 4;        // 512 KiB
constexpr size_t OFF_H     = OFF_RSTD  + (size_t)B_ * N_ * 4;        // 2 MiB
constexpr size_t OFF_H2    = OFF_H     + (size_t)B_ * C_ * S_ * 4;   // 1 MiB

extern "C" void kernel_launch(void* const* d_in, const int* in_sizes, int n_in,
                              void* d_out, int out_size, void* d_ws, size_t ws_size,
                              hipStream_t stream) {
  (void)in_sizes; (void)n_in; (void)out_size; (void)ws_size;
  const float* x   = (const float*)d_in[0];
  const float* lnw = (const float*)d_in[1];
  const float* lnb = (const float*)d_in[2];
  const float* wb  = (const float*)d_in[3];
  const float* wg  = (const float*)d_in[4];
  const float* wo  = (const float*)d_in[5];
  const float* A   = (const float*)d_in[6];
  float* out = (float*)d_out;

  char* ws = (char*)d_ws;
  u16*   xbf   = (u16*)(ws + OFF_XBF);
  u16*   bcdt  = (u16*)(ws + OFF_BCDT);
  u16*   bcdt2 = (u16*)(ws + OFF_BCDT2);
  u16*   ab    = (u16*)(ws + OFF_AB);
  float* mean  = (float*)(ws + OFF_MEAN);
  float* rstd  = (float*)(ws + OFF_RSTD);
  float* h     = (float*)(ws + OFF_H);
  u16*   h2    = (u16*)(ws + OFF_H2);

  k_ln        <<<dim3((B_ * N_) / 256), 256, 0, stream>>>(x, mean, rstd, xbf);
  k_bcdt_gemm <<<dim3(N_ / 64, CD_ / 64, B_), 256, 0, stream>>>(xbf, wb, mean, rstd, lnw, lnb, bcdt);
  k_gconv     <<<dim3(N_ / 256, CD_, B_), 256, 0, stream>>>(bcdt, wg, bcdt2);
  k_softmax_ab<<<dim3(S_, B_), 256, 0, stream>>>(bcdt2, A, ab);
  k_h_gemm    <<<dim3(C_ / 64, B_), 256, 0, stream>>>(xbf, ab, h);
  k_h2        <<<dim3(C_ / 4, B_), 256, 0, stream>>>(wo, h, h2);
  k_y_out     <<<dim3(N_ / 64, C_ / 64, B_), 256, 0, stream>>>(h2, bcdt2, x, out);
}
